// Block_77386720739529
// MI455X (gfx1250) — matrix operand order repaired and checked by reading, not compiled, not run on hardware
//
#include <hip/hip_runtime.h>

// ---------------------------------------------------------------- types ----
typedef __bf16 bf16;
typedef __attribute__((ext_vector_type(8)))  __bf16 v8bf;
typedef __attribute__((ext_vector_type(16))) __bf16 v16bf;
typedef __attribute__((ext_vector_type(8)))  float  v8f;
typedef unsigned int u32x4 __attribute__((ext_vector_type(4)));
typedef int i32x4 __attribute__((ext_vector_type(4)));
typedef int i32x8 __attribute__((ext_vector_type(8)));

#define DEV __device__ __forceinline__

// Tensor Data Mover availability (device pass only; host pass sees 0)
#ifndef USE_TDM
#  if defined(__has_builtin)
#    if __has_builtin(__builtin_amdgcn_tensor_load_to_lds) && \
        __has_builtin(__builtin_amdgcn_s_wait_tensorcnt)
#      define USE_TDM 1
#    else
#      define USE_TDM 0
#    endif
#  else
#    define USE_TDM 0
#  endif
#endif

DEV bf16 f2bf(float f) {
    unsigned u = __builtin_bit_cast(unsigned, f);
    unsigned r = u + 0x7FFFu + ((u >> 16) & 1u);   // round-to-nearest-even
    unsigned short h = (unsigned short)(r >> 16);
    return __builtin_bit_cast(bf16, h);
}

DEV v16bf combine8(v8bf lo, v8bf hi) {
    return __builtin_shufflevector(lo, hi, 0,1,2,3,4,5,6,7,8,9,10,11,12,13,14,15);
}

DEV v8f v8f_zero() {
    v8f z;
#pragma unroll
    for (int i = 0; i < 8; ++i) z[i] = 0.0f;
    return z;
}

DEV v8f wmma_bf16(v16bf a, v16bf b, v8f c) {
    // D = A(16x32) * B(32x16) + C, fp32 accumulate
    return __builtin_amdgcn_wmma_f32_16x16x32_bf16(false, a, false, b, (short)0, c, false, false);
}

// model dims
static constexpr int T     = 2048;
static constexpr int CDIM  = 2048;
static constexpr int HEADS = 16;
static constexpr int DK    = 128;
static constexpr int FF    = 8192;
static constexpr int MROWS = 2 * T;   // B*T = 4096

// ---------------------------------------------------------------- TDM ------
#if USE_TDM
// Issue a TDM load of a 128-row x 64-elem bf16 tile (row stride = kTot elems)
// into LDS with 16B padding after each 128B row (LDS row stride 144B).
DEV void tdm_issue_tile(bf16* ldsDst, const bf16* gsrc, int nRows, int kTot) {
    unsigned long long ga = (unsigned long long)(const void*)gsrc;
    u32x4 g0;
    g0[0] = 1u;                                            // count=1, user desc
    g0[1] = (unsigned)(unsigned long long)(void*)ldsDst;   // lds_addr (bytes)
    g0[2] = (unsigned)(ga & 0xFFFFFFFFull);                // global_addr[31:0]
    g0[3] = (unsigned)((ga >> 32) & 0x01FFFFFFull) | 0x80000000u; // [56:32] | type=2
    i32x8 g1;
    // data_size=2B, pad_enable, pad_interval=32 DWORDs(code 4), pad_amount=4 DWORDs(code 3)
    g1[0] = (int)((1u << 16) | (1u << 20) | (4u << 22) | (3u << 25));
    g1[1] = (int)(((unsigned)kTot & 0xFFFFu) << 16);       // tensor_dim0 lo16
    g1[2] = (int)(((unsigned)kTot >> 16) | (((unsigned)nRows & 0xFFFFu) << 16));
    g1[3] = (int)(((unsigned)nRows >> 16) | (64u << 16));  // dim1 hi | tile_dim0=64
    g1[4] = 128;                                           // tile_dim1=128, tile_dim2=0
    g1[5] = (int)(unsigned)kTot;                           // tensor_dim0_stride lo32
    g1[6] = 0;
    g1[7] = 0;
    i32x4 gz;
    gz[0] = 0; gz[1] = 0; gz[2] = 0; gz[3] = 0;
#if __clang_major__ >= 23
    i32x8 gz8;
#pragma unroll
    for (int i = 0; i < 8; ++i) gz8[i] = 0;
    __builtin_amdgcn_tensor_load_to_lds(g0, g1, gz, gz, gz8, 0);
#else
    __builtin_amdgcn_tensor_load_to_lds(g0, g1, gz, gz, 0);
#endif
}
#endif

// ------------------------------------------------------------- convert ----
__global__ void cvt_f32_bf16(const float* __restrict__ src, bf16* __restrict__ dst, int n) {
    int i = blockIdx.x * blockDim.x + threadIdx.x;
    int stride = gridDim.x * blockDim.x;
    for (; i < n; i += stride) dst[i] = f2bf(src[i]);
}

// ------------------------------------------------------------- rmsnorm ----
__launch_bounds__(256)
__global__ void rmsnorm_kernel(const float* __restrict__ x, const float* __restrict__ g,
                               bf16* __restrict__ out) {
    __shared__ float red[8];
    const int row = blockIdx.x;
    const float* xr = x + (size_t)row * CDIM;
    float vals[8];
    float ss = 0.0f;
#pragma unroll
    for (int i = 0; i < 8; ++i) {
        float v = xr[threadIdx.x + i * 256];
        vals[i] = v;
        ss += v * v;
    }
#pragma unroll
    for (int m = 16; m >= 1; m >>= 1) ss += __shfl_xor(ss, m, 32);
    if ((threadIdx.x & 31) == 0) red[threadIdx.x >> 5] = ss;
    __syncthreads();
    float tot = 0.0f;
#pragma unroll
    for (int i = 0; i < 8; ++i) tot += red[i];
    const float norm = rsqrtf(tot / (float)CDIM + 1e-6f);
#pragma unroll
    for (int i = 0; i < 8; ++i) {
        int c = threadIdx.x + i * 256;
        out[(size_t)row * CDIM + c] = f2bf(vals[i] * norm * g[c]);
    }
}

// ----------------------------------------------------------------- gemm ----
// C[M,N] = A[M,K](bf16) * W[N,K]^T(bf16), fp32 accum via v_wmma_f32_16x16x32_bf16.
// block = 256 thr (8 waves), block tile 128x128, wave tile 32x64 (2x4 frags).
// Weight tiles (128x64 bf16) are staged into LDS by the Tensor Data Mover,
// double buffered: wave 0 issues the next tile's TDM DMA, then the whole block
// computes WMMAs out of the current LDS buffer while the DMA flies.
// EPI 0: outF = acc (+ res if res!=null);  EPI 1: outB = bf16(silu(other)*acc)
template <int EPI>
__launch_bounds__(256)
__global__ void gemm_bf16_kernel(const bf16* __restrict__ A, const bf16* __restrict__ W,
                                 int M, int N, int K,
                                 const float* __restrict__ res,
                                 const float* __restrict__ other,
                                 float* __restrict__ outF,
                                 bf16*  __restrict__ outB) {
    constexpr int TK   = 64;   // K elems staged per tile
    constexpr int LSTR = 72;   // padded LDS row stride in elems (144B = 36 banks)
    __shared__ bf16 smw[2][128 * LSTR];

    const int lane = threadIdx.x & 31;
    const int wid  = threadIdx.x >> 5;
    const int bM  = blockIdx.y * 128 + (wid & 3) * 32;   // 4 wave-rows
    const int bN0 = blockIdx.x * 128;
    const int wn  = (wid >> 2) * 64;                     // 2 wave-cols
    const int mA = lane & 15, kh = (lane >> 4) * 16;     // A-fragment layout
    const int nB = lane & 15, kb = (lane >> 4) * 16;     // B-fragment layout

    const bf16* arow[2];
#pragma unroll
    for (int mi = 0; mi < 2; ++mi) arow[mi] = A + (size_t)(bM + mi * 16 + mA) * K;
    const bf16* wblk = W + (size_t)bN0 * K;

    v8f acc[2][4];
#pragma unroll
    for (int mi = 0; mi < 2; ++mi)
#pragma unroll
        for (int ni = 0; ni < 4; ++ni) acc[mi][ni] = v8f_zero();

    const int nk = K / TK;

#if USE_TDM
    if (wid == 0) tdm_issue_tile(&smw[0][0], wblk, N, K);
#endif

    for (int kt = 0; kt < nk; ++kt) {
        const int k0 = kt * TK;
        bf16* sw = &smw[kt & 1][0];
#if USE_TDM
        if (wid == 0) {
            if (kt + 1 < nk) {
                tdm_issue_tile(&smw[(kt + 1) & 1][0], wblk + (k0 + TK), N, K);
                __builtin_amdgcn_s_wait_tensorcnt(1);   // tile kt landed
            } else {
                __builtin_amdgcn_s_wait_tensorcnt(0);
            }
        }
        __syncthreads();
#else
        // cooperative staged copy with the same padded layout
        for (int idx = threadIdx.x; idx < 128 * 4; idx += 256) {
            const int row = idx >> 2, seg = idx & 3;
            *(v16bf*)(sw + row * LSTR + seg * 16) =
                *(const v16bf*)(wblk + (size_t)row * K + k0 + seg * 16);
        }
        __syncthreads();
#endif
        // A fragments for both 32-K half-steps up front (overlaps with LDS reads)
        v16bf a[2][2];
#pragma unroll
        for (int mi = 0; mi < 2; ++mi)
#pragma unroll
            for (int ks = 0; ks < 2; ++ks) {
                const bf16* p = arow[mi] + k0 + ks * 32;
                a[mi][ks] = combine8(*(const v8bf*)(p + kh), *(const v8bf*)(p + kh + 8));
            }
        __builtin_prefetch((const void*)(arow[0] + k0 + TK), 0, 1);
#pragma unroll
        for (int ks = 0; ks < 2; ++ks) {
            v16bf b[4];
#pragma unroll
            for (int ni = 0; ni < 4; ++ni)
                b[ni] = *(const v16bf*)(sw + (wn + ni * 16 + nB) * LSTR + ks * 32 + kb);
#pragma unroll
            for (int mi = 0; mi < 2; ++mi)
#pragma unroll
                for (int ni = 0; ni < 4; ++ni)
                    acc[mi][ni] = wmma_bf16(a[mi][ks], b[ni], acc[mi][ni]);
        }
        __syncthreads();   // all reads done before next tile overwrites this buffer
    }

    const int rb = (lane >> 4) * 8, nc = lane & 15;      // C-fragment layout
#pragma unroll
    for (int mi = 0; mi < 2; ++mi)
#pragma unroll
        for (int ni = 0; ni < 4; ++ni)
#pragma unroll
            for (int r = 0; r < 8; ++r) {
                const int row = bM + mi * 16 + rb + r;
                const int col = bN0 + wn + ni * 16 + nc;
                const size_t idx = (size_t)row * N + col;
                float v = acc[mi][ni][r];
                if (EPI == 0) {
                    if (res) v += res[idx];
                    outF[idx] = v;
                } else {
                    float gt = other[idx];
                    float sl = gt / (1.0f + __expf(-gt));   // silu(gate)
                    outB[idx] = f2bf(sl * v);
                }
            }
}

// ----------------------------------------------------------------- rope ----
__launch_bounds__(256)
__global__ void rope_kernel(const float* __restrict__ qkv,
                            const float* __restrict__ fcos, const float* __restrict__ fsin,
                            bf16* __restrict__ qb, bf16* __restrict__ kb,
                            bf16* __restrict__ vt) {
    const int row = blockIdx.x;          // b*T + t
    const int b = row >> 11, t = row & (T - 1);
    const float* qr = qkv + (size_t)row * (3 * CDIM);
    const float scale = 0.08838834764831845f;   // 1/sqrt(128)
    for (int idx = threadIdx.x; idx < HEADS * (DK / 2); idx += 256) {
        const int h = idx >> 6, i = idx & 63;
        const float c = fcos[t * 64 + i], s = fsin[t * 64 + i];
        const size_t base = ((size_t)(b * HEADS + h) * T + t) * DK;
        float xr = qr[h * DK + 2 * i], xi = qr[h * DK + 2 * i + 1];
        qb[base + 2 * i]     = f2bf((xr * c - xi * s) * scale);
        qb[base + 2 * i + 1] = f2bf((xr * s + xi * c) * scale);
        float yr = qr[CDIM + h * DK + 2 * i], yi = qr[CDIM + h * DK + 2 * i + 1];
        kb[base + 2 * i]     = f2bf(yr * c - yi * s);
        kb[base + 2 * i + 1] = f2bf(yr * s + yi * c);
    }
    for (int idx = threadIdx.x; idx < HEADS * DK; idx += 256) {
        const int h = idx >> 7, d = idx & (DK - 1);
        float v = qr[2 * CDIM + h * DK + d];
        vt[((size_t)(b * HEADS + h) * DK + d) * T + t] = f2bf(v);
    }
}

// ------------------------------------------------------------ attention ----
// flash-style causal attention. 1 wave / block; block handles a 32-row q tile
// of one (b,h). S = Q K^T and O += P V via WMMA; P goes through LDS to change
// from C-layout to A-layout. All fragment loads for a KV step are hoisted into
// one clause so a single s_wait covers them before the WMMA burst.
__launch_bounds__(32)
__global__ void attn_kernel(const bf16* __restrict__ qb, const bf16* __restrict__ kq,
                            const bf16* __restrict__ vt, bf16* __restrict__ outb) {
    __shared__ bf16 pls[32 * 40];                 // 32x32 P tile, padded stride 40
    const int lane = threadIdx.x;
    const int qt = blockIdx.x, bh = blockIdx.y;
    const int b = bh >> 4, h = bh & 15;
    const int q0 = qt * 32;
    const int mA = lane & 15, kh = (lane >> 4) * 16;
    const int nB = lane & 15, kbo = (lane >> 4) * 16;
    const int rb = (lane >> 4) * 8, nc = lane & 15;

    v16bf qf[2][4];
#pragma unroll
    for (int mi = 0; mi < 2; ++mi)
#pragma unroll
        for (int kk = 0; kk < 4; ++kk) {
            const bf16* p = qb + ((size_t)bh * T + q0 + mi * 16 + mA) * DK + kk * 32;
            qf[mi][kk] = combine8(*(const v8bf*)(p + kh), *(const v8bf*)(p + kh + 8));
        }

    v8f o[2][8];
#pragma unroll
    for (int mi = 0; mi < 2; ++mi)
#pragma unroll
        for (int nj = 0; nj < 8; ++nj) o[mi][nj] = v8f_zero();
    float rmax[2][8], rsum[2][8];
#pragma unroll
    for (int mi = 0; mi < 2; ++mi)
#pragma unroll
        for (int r = 0; r < 8; ++r) { rmax[mi][r] = -3.0e38f; rsum[mi][r] = 0.0f; }

    for (int j = 0; j <= qt; ++j) {
        const int k0 = j * 32;
        // hoist ALL loads of this KV step: 8 K-fragments + 8 V-fragments
        v16bf kf[2][4];
#pragma unroll
        for (int ni = 0; ni < 2; ++ni)
#pragma unroll
            for (int kk = 0; kk < 4; ++kk) {
                const bf16* p = kq + ((size_t)bh * T + k0 + ni * 16 + nB) * DK + kk * 32 + kbo;
                kf[ni][kk] = *(const v16bf*)p;
            }
        v16bf vf[8];
#pragma unroll
        for (int nj = 0; nj < 8; ++nj) {
            const bf16* vp = vt + ((size_t)bh * DK + nj * 16 + nB) * T + k0 + kbo;
            vf[nj] = *(const v16bf*)vp;
        }

        v8f s[2][2];
#pragma unroll
        for (int mi = 0; mi < 2; ++mi)
#pragma unroll
            for (int ni = 0; ni < 2; ++ni) {
                s[mi][ni] = v8f_zero();
#pragma unroll
                for (int kk = 0; kk < 4; ++kk)
                    s[mi][ni] = wmma_bf16(qf[mi][kk], kf[ni][kk], s[mi][ni]);
            }
        if (j == qt) {   // causal mask only on the diagonal tile
#pragma unroll
            for (int mi = 0; mi < 2; ++mi)
#pragma unroll
                for (int ni = 0; ni < 2; ++ni)
#pragma unroll
                    for (int r = 0; r < 8; ++r) {
                        const int rowi = mi * 16 + rb + r;
                        const int keyi = ni * 16 + nc;
                        s[mi][ni][r] = (keyi > rowi) ? -3.0e38f : s[mi][ni][r];
                    }
        }
        float pex[2][2][8];
#pragma unroll
        for (int mi = 0; mi < 2; ++mi)
#pragma unroll
            for (int r = 0; r < 8; ++r) {
                float tmax = fmaxf(s[mi][0][r], s[mi][1][r]);
#pragma unroll
                for (int m = 8; m >= 1; m >>= 1) tmax = fmaxf(tmax, __shfl_xor(tmax, m, 16));
                const float nm = fmaxf(rmax[mi][r], tmax);
                const float al = __expf(rmax[mi][r] - nm);
                rmax[mi][r] = nm;
                float p0 = __expf(s[mi][0][r] - nm);
                float p1 = __expf(s[mi][1][r] - nm);
                pex[mi][0][r] = p0;
                pex[mi][1][r] = p1;
                float ps = p0 + p1;
#pragma unroll
                for (int m = 8; m >= 1; m >>= 1) ps += __shfl_xor(ps, m, 16);
                rsum[mi][r] = rsum[mi][r] * al + ps;
#pragma unroll
                for (int nj = 0; nj < 8; ++nj) o[mi][nj][r] *= al;
            }
#pragma unroll
        for (int mi = 0; mi < 2; ++mi)
#pragma unroll
            for (int ni = 0; ni < 2; ++ni)
#pragma unroll
                for (int r = 0; r < 8; ++r)
                    pls[(mi * 16 + rb + r) * 40 + ni * 16 + nc] = f2bf(pex[mi][ni][r]);
        __syncthreads();
        v16bf pa[2];
#pragma unroll
        for (int mi = 0; mi < 2; ++mi)
            pa[mi] = combine8(*(const v8bf*)(pls + (mi * 16 + mA) * 40 + kh),
                              *(const v8bf*)(pls + (mi * 16 + mA) * 40 + kh + 8));
        // 16 back-to-back WMMAs; V fragments already resident
#pragma unroll
        for (int nj = 0; nj < 8; ++nj)
#pragma unroll
            for (int mi = 0; mi < 2; ++mi)
                o[mi][nj] = wmma_bf16(pa[mi], vf[nj], o[mi][nj]);
        __syncthreads();
    }

#pragma unroll
    for (int mi = 0; mi < 2; ++mi)
#pragma unroll
        for (int r = 0; r < 8; ++r) {
            const float inv = 1.0f / rsum[mi][r];
            const int t = q0 + mi * 16 + rb + r;
#pragma unroll
            for (int nj = 0; nj < 8; ++nj) {
                const int col = h * DK + nj * 16 + nc;
                outb[((size_t)b * T + t) * CDIM + col] = f2bf(o[mi][nj][r] * inv);
            }
        }
}

// --------------------------------------------------------------- launch ----
extern "C" void kernel_launch(void* const* d_in, const int* in_sizes, int n_in,
                              void* d_out, int out_size, void* d_ws, size_t ws_size,
                              hipStream_t stream) {
    const float* x      = (const float*)d_in[0];
    const float* fcos   = (const float*)d_in[1];
    const float* fsin   = (const float*)d_in[2];
    const float* w_attn = (const float*)d_in[3];
    const float* w_proj = (const float*)d_in[4];
    const float* w_gate = (const float*)d_in[5];
    const float* w_up   = (const float*)d_in[6];
    const float* w_down = (const float*)d_in[7];
    const float* g1     = (const float*)d_in[8];
    const float* g2     = (const float*)d_in[9];
    float* out = (float*)d_out;

    char* ws = (char*)d_ws;
    size_t off = 0;
    auto take = [&](size_t bytes) -> char* {
        char* p = ws + off;
        off = (off + bytes + 255) & ~(size_t)255;
        return p;
    };

    const size_t N_ATTN = (size_t)3 * CDIM * CDIM;
    const size_t N_PROJ = (size_t)CDIM * CDIM;
    const size_t N_FFW  = (size_t)FF * CDIM;
    const size_t N_ACT  = (size_t)MROWS * CDIM;

    bf16* wb_attn = (bf16*)take(N_ATTN * 2);
    bf16* wb_proj = (bf16*)take(N_PROJ * 2);
    bf16* wb_gate = (bf16*)take(N_FFW * 2);
    bf16* wb_up   = (bf16*)take(N_FFW * 2);
    bf16* wb_down = (bf16*)take(N_FFW * 2);
    bf16* h_bf    = (bf16*)take(N_ACT * 2);
    char* qkv_region = take((size_t)MROWS * 3 * CDIM * 4);   // 100 MB, reused below
    float* qkvf = (float*)qkv_region;
    bf16*  m_bf = (bf16*)qkv_region;                         // M*FF bf16 = 67 MB fits
    bf16* q_bf  = (bf16*)take(N_ACT * 2);
    bf16* k_bf  = (bf16*)take(N_ACT * 2);
    bf16* vt_bf = (bf16*)take(N_ACT * 2);
    bf16* attn_bf = (bf16*)take(N_ACT * 2);
    float* x2   = (float*)take(N_ACT * 4);
    float* gatef = (float*)take((size_t)MROWS * FF * 4);

    // weights -> bf16 (bf16 weight set = 134 MB, resident in 192 MB L2)
    cvt_f32_bf16<<<2048, 256, 0, stream>>>(w_attn, wb_attn, (int)N_ATTN);
    cvt_f32_bf16<<<2048, 256, 0, stream>>>(w_proj, wb_proj, (int)N_PROJ);
    cvt_f32_bf16<<<2048, 256, 0, stream>>>(w_gate, wb_gate, (int)N_FFW);
    cvt_f32_bf16<<<2048, 256, 0, stream>>>(w_up,   wb_up,   (int)N_FFW);
    cvt_f32_bf16<<<2048, 256, 0, stream>>>(w_down, wb_down, (int)N_FFW);

    // h = rmsnorm(x, g1)
    rmsnorm_kernel<<<MROWS, 256, 0, stream>>>(x, g1, h_bf);

    // qkv = h @ w_attn^T   [4096 x 6144]
    gemm_bf16_kernel<0><<<dim3(6144 / 128, MROWS / 128), 256, 0, stream>>>(
        h_bf, wb_attn, MROWS, 3 * CDIM, CDIM, nullptr, nullptr, qkvf, nullptr);

    // rope + head split + V transpose
    rope_kernel<<<MROWS, 256, 0, stream>>>(qkvf, fcos, fsin, q_bf, k_bf, vt_bf);

    // causal flash attention
    attn_kernel<<<dim3(T / 32, 2 * HEADS), 32, 0, stream>>>(q_bf, k_bf, vt_bf, attn_bf);

    // x2 = x + attn @ w_proj^T
    gemm_bf16_kernel<0><<<dim3(CDIM / 128, MROWS / 128), 256, 0, stream>>>(
        attn_bf, wb_proj, MROWS, CDIM, CDIM, x, nullptr, x2, nullptr);

    // h2 = rmsnorm(x2, g2)
    rmsnorm_kernel<<<MROWS, 256, 0, stream>>>(x2, g2, h_bf);

    // gate = h2 @ w_gate^T   [4096 x 8192]
    gemm_bf16_kernel<0><<<dim3(FF / 128, MROWS / 128), 256, 0, stream>>>(
        h_bf, wb_gate, MROWS, FF, CDIM, nullptr, nullptr, gatef, nullptr);

    // m = silu(gate) * (h2 @ w_up^T)  -> bf16
    gemm_bf16_kernel<1><<<dim3(FF / 128, MROWS / 128), 256, 0, stream>>>(
        h_bf, wb_up, MROWS, FF, CDIM, nullptr, gatef, nullptr, m_bf);

    // out = x2 + m @ w_down^T
    gemm_bf16_kernel<0><<<dim3(CDIM / 128, MROWS / 128), 256, 0, stream>>>(
        m_bf, wb_down, MROWS, CDIM, FF, x2, nullptr, out, nullptr);

    (void)in_sizes; (void)n_in; (void)out_size; (void)ws_size;
}
